// KANLinear_18743237280423
// MI455X (gfx1250) — compile-verified, matrix operand order checked by reading, not run on hardware
//
#include <hip/hip_runtime.h>

// ---------------------------------------------------------------------------
// KAN linear layer, fused as one augmented bf16 GEMM on gfx1250 WMMA.
//   out[b][o] = sum_k A[b][k] * B[o][k],  K = 1024 (x part) + 8192 (spline part)
//   A[b][k]        = x[b][k]                          (k < 1024)
//   A[b][1024+i*8+j] = basis_j(x[b][i])               (cubic B-spline, on the fly)
//   B[o][k]        = base_weight[o][k]                (k < 1024)
//   B[o][1024+i*8+j] = spline_weight[o][i][j]*scaler[o][i]   (packed once to ws)
// ---------------------------------------------------------------------------

typedef __attribute__((ext_vector_type(16))) __bf16 v16bf;
typedef __attribute__((ext_vector_type(8)))  __bf16 v8bf;
typedef __attribute__((ext_vector_type(4)))  __bf16 v4bf;
typedef __attribute__((ext_vector_type(8)))  float  v8f;

#define IN_FEATURES  1024
#define OUT_FEATURES 1024
#define NUM_BASIS    8
#define KDIM         (IN_FEATURES + IN_FEATURES * NUM_BASIS)   // 9216
#define N_TOKENS     8192

#define BM  128      // token tile
#define BN  128      // output tile
#define KB  32       // K step (one bf16 WMMA K)
#define LDT 40       // padded LDS row stride in elements (80 B = 5 x 16 B)

// float -> bf16, round-to-nearest-even
__device__ __forceinline__ __bf16 f2bf(float f) {
    union { float f; unsigned int u; } v; v.f = f;
    unsigned int r = v.u + 0x7FFFu + ((v.u >> 16) & 1u);
    union { unsigned short u; __bf16 b; } o; o.u = (unsigned short)(r >> 16);
    return o.b;
}
__device__ __forceinline__ unsigned short f2bfu(float f) {
    union { float f; unsigned int u; } v; v.f = f;
    unsigned int r = v.u + 0x7FFFu + ((v.u >> 16) & 1u);
    return (unsigned short)(r >> 16);
}

// Cubic B-spline bases on uniform grid, matching the reference recurrence:
//   b_new[j] = ((x - g[k+j]) * b[j] + (g[k+1+j] - x) * b[j+1]) / (k*h)
__device__ __forceinline__ void bspline8(float x, float b[8]) {
    const float h = 0.4f;                      // (1 - (-1)) / GRID_SIZE
    float g[12];
#pragma unroll
    for (int j = 0; j < 12; ++j) g[j] = -1.0f + h * (float)(j - 3);
    float t0[11];
#pragma unroll
    for (int j = 0; j < 11; ++j) t0[j] = (x >= g[j] && x < g[j + 1]) ? 1.0f : 0.0f;
    float t1[10];
#pragma unroll
    for (int j = 0; j < 10; ++j)
        t1[j] = ((x - g[j + 1]) * t0[j] + (g[j + 2] - x) * t0[j + 1]) * (1.0f / h);
    float t2[9];
#pragma unroll
    for (int j = 0; j < 9; ++j)
        t2[j] = ((x - g[j + 2]) * t1[j] + (g[j + 3] - x) * t1[j + 1]) * (1.0f / (2.0f * h));
#pragma unroll
    for (int j = 0; j < 8; ++j)
        b[j] = ((x - g[j + 3]) * t2[j] + (g[j + 4] - x) * t2[j + 1]) * (1.0f / (3.0f * h));
}

// ---------------------------------------------------------------------------
// Kernel 1: pack [base_weight | spline_weight*scaler] into bf16 Bpack[o][KDIM]
// ---------------------------------------------------------------------------
__global__ __launch_bounds__(256) void kan_pack_b(
    const float* __restrict__ base_w,
    const float* __restrict__ spline_w,
    const float* __restrict__ scaler,
    unsigned short* __restrict__ bpack) {
    int idx = blockIdx.x * 256 + threadIdx.x;
    if (idx >= OUT_FEATURES * KDIM) return;
    int o = idx / KDIM;
    int k = idx - o * KDIM;
    float v;
    if (k < IN_FEATURES) {
        v = base_w[o * IN_FEATURES + k];
    } else {
        int s = k - IN_FEATURES;
        int i = s >> 3;
        int j = s & 7;
        v = spline_w[(o * IN_FEATURES + i) * NUM_BASIS + j] * scaler[o * IN_FEATURES + i];
    }
    bpack[o * KDIM + k] = f2bfu(v);
}

// ---------------------------------------------------------------------------
// Kernel 2: fused GEMM. Grid = (8192/128, 1024/128), 256 thr = 8 wave32.
// Each wave: 32x64 sub-tile = 2x4 v_wmma_f32_16x16x32_bf16 per K step.
// ---------------------------------------------------------------------------
__global__ __launch_bounds__(256) void kan_fused_gemm(
    const float* __restrict__ x,
    const unsigned short* __restrict__ bpack,
    float* __restrict__ out) {

    __shared__ __bf16 As[BM * LDT];   // token rows  x K-slice (bf16)
    __shared__ __bf16 Bs[BN * LDT];   // output rows x K-slice (bf16)

    const int tid  = threadIdx.x;
    const int lane = tid & 31;
    const int wave = tid >> 5;
    const int wm   = wave & 3;              // M offset = wm*32
    const int wn   = wave >> 2;             // N offset = wn*64
    const int tok_base = blockIdx.x * BM;
    const int out_base = blockIdx.y * BN;

    const int lhalf = lane & 15;
    const int k0    = (lane & 16) ? 8 : 0;  // per-lane K base within fragment

    v8f acc[2][4];
#pragma unroll
    for (int mi = 0; mi < 2; ++mi)
#pragma unroll
        for (int ni = 0; ni < 4; ++ni) acc[mi][ni] = (v8f)(0.0f);

    for (int kb = 0; kb < KDIM; kb += KB) {
        // ---- stage B tile: 128 x 32 bf16 from packed weights (L2 resident) ----
#pragma unroll
        for (int p = 0; p < 2; ++p) {
            int c    = tid + p * 256;       // 0..511 chunks of 16 B
            int n    = c >> 2;
            int part = c & 3;
            v8bf bv = *(const v8bf*)(bpack + (out_base + n) * KDIM + kb + part * 8);
            *(v8bf*)(&Bs[n * LDT + part * 8]) = bv;
        }
        // ---- stage A tile ----
        if (kb < IN_FEATURES) {
            // x part: 128 rows x 32 features, f32 -> bf16
            int c4 = tid & 7;               // column group of 4
            int r0 = tid >> 3;              // 0..31
#pragma unroll
            for (int rr = 0; rr < 4; ++rr) {
                int row = r0 + rr * 32;
                float4 xv = *(const float4*)(x + (size_t)(tok_base + row) * IN_FEATURES + kb + c4 * 4);
                v4bf bv;
                bv[0] = f2bf(xv.x); bv[1] = f2bf(xv.y);
                bv[2] = f2bf(xv.z); bv[3] = f2bf(xv.w);
                *(v4bf*)(&As[row * LDT + c4 * 4]) = bv;
            }
        } else {
            // spline part: this K slice covers 4 input features x 8 bases
            int ibase = (kb - IN_FEATURES) >> 3;
            int f  = tid & 3;               // local feature 0..3
            int r0 = tid >> 2;              // 0..63
#pragma unroll
            for (int rr = 0; rr < 2; ++rr) {
                int row = r0 + rr * 64;
                float xv = x[(size_t)(tok_base + row) * IN_FEATURES + ibase + f];
                float bas[8];
                bspline8(xv, bas);
                v8bf bv;
#pragma unroll
                for (int j = 0; j < 8; ++j) bv[j] = f2bf(bas[j]);
                *(v8bf*)(&As[row * LDT + f * 8]) = bv;
            }
        }
        __syncthreads();

        // ---- fragments straight from LDS in the ISA 16-bit A/B layout ----
        v16bf afr[2];
#pragma unroll
        for (int mi = 0; mi < 2; ++mi) {
            const __bf16* base = &As[(wm * 32 + mi * 16 + lhalf) * LDT + k0];
            v8bf lo = *(const v8bf*)base;
            v8bf hi = *(const v8bf*)(base + 16);
#pragma unroll
            for (int j = 0; j < 8; ++j) { afr[mi][j] = lo[j]; afr[mi][8 + j] = hi[j]; }
        }
        v16bf bfr[4];
#pragma unroll
        for (int ni = 0; ni < 4; ++ni) {
            const __bf16* base = &Bs[(wn * 64 + ni * 16 + lhalf) * LDT + k0];
            v8bf lo = *(const v8bf*)base;
            v8bf hi = *(const v8bf*)(base + 16);
#pragma unroll
            for (int j = 0; j < 8; ++j) { bfr[ni][j] = lo[j]; bfr[ni][8 + j] = hi[j]; }
        }
#pragma unroll
        for (int mi = 0; mi < 2; ++mi)
#pragma unroll
            for (int ni = 0; ni < 4; ++ni)
                acc[mi][ni] = __builtin_amdgcn_wmma_f32_16x16x32_bf16(
                    false, afr[mi], false, bfr[ni],
                    (short)0, acc[mi][ni], false, false);
        __syncthreads();
    }

    // ---- epilogue: f32 C layout (VGPR r: lanes 0-15 M=r, lanes 16-31 M=8+r) ----
    const int rbase = (lane & 16) ? 8 : 0;
#pragma unroll
    for (int mi = 0; mi < 2; ++mi)
#pragma unroll
        for (int ni = 0; ni < 4; ++ni) {
            int col = out_base + wn * 64 + ni * 16 + lhalf;
#pragma unroll
            for (int r = 0; r < 8; ++r) {
                int row = tok_base + wm * 32 + mi * 16 + rbase + r;
                out[(size_t)row * OUT_FEATURES + col] = acc[mi][ni][r];
            }
        }
}

// ---------------------------------------------------------------------------
extern "C" void kernel_launch(void* const* d_in, const int* in_sizes, int n_in,
                              void* d_out, int out_size, void* d_ws, size_t ws_size,
                              hipStream_t stream) {
    (void)in_sizes; (void)n_in; (void)out_size; (void)ws_size;
    const float* x        = (const float*)d_in[0];
    const float* base_w   = (const float*)d_in[1];
    const float* spline_w = (const float*)d_in[2];
    const float* scaler   = (const float*)d_in[3];
    float* out            = (float*)d_out;
    unsigned short* bpack = (unsigned short*)d_ws;   // 1024*9216*2 B = 18.9 MB

    int total = OUT_FEATURES * KDIM;
    kan_pack_b<<<(total + 255) / 256, 256, 0, stream>>>(base_w, spline_w, scaler, bpack);

    dim3 grid(N_TOKENS / BM, OUT_FEATURES / BN);
    kan_fused_gemm<<<grid, 256, 0, stream>>>(x, bpack, out);
}
